// HAWPv1_10960756539487
// MI455X (gfx1250) — matrix operand order repaired and checked by reading, not compiled, not run on hardware
//
#include <hip/hip_runtime.h>

#define HWD   16384        // H*W
#define Wd    128
#define Hd    128
#define NL    49152        // 3*H*W lines
#define KJ    300
#define SENT  (KJ*KJ)      // 90000
#define PI_F  3.14159265358979f

typedef __attribute__((ext_vector_type(16))) __bf16   v16bf;
typedef __attribute__((ext_vector_type(8)))  float    v8f;
typedef __attribute__((ext_vector_type(4)))  unsigned u32x4;
typedef __attribute__((ext_vector_type(8)))  int      i32x8;
typedef __attribute__((ext_vector_type(4)))  int      i32x4;

union Frag16 { uint4 q[2]; v16bf v; };   // two b128 loads -> one WMMA operand

#if __has_builtin(__builtin_amdgcn_tensor_load_to_lds)
#define HAVE_TDM 1
#else
#define HAVE_TDM 0
#endif

__device__ __forceinline__ float sigmoidf(float x){ return 1.f/(1.f+expf(-x)); }

#if HAVE_TDM
// TDM descriptor (ISA cdna5 §8): 2-D tensor of bf16, row length 1024 elems,
// tile = 32(K, contiguous) x 1024(N rows), stride 1024 -> 64KB slab into LDS.
__device__ __forceinline__ void tdm_slab_load(unsigned lds_off, const __bf16* gsrc){
  unsigned long long ga = (unsigned long long)(size_t)gsrc;
  u32x4 g0;
  g0[0] = 1u;                                            // count=1, user mode
  g0[1] = lds_off;                                       // lds_addr (bytes)
  g0[2] = (unsigned)ga;                                  // global_addr lo
  g0[3] = (unsigned)((ga >> 32) & 0x01FFFFFFull) | (2u << 30); // addr hi | type=2
  i32x8 g1;
  g1[0] = 1 << 16;                 // data_size = 1 (2 bytes/elem)
  g1[1] = (int)(1024u << 16);      // tensor_dim0[15:0]=1024 at bits[31:16]
  g1[2] = (int)(1024u << 16);      // dim0 hi=0 | tensor_dim1[15:0]=1024
  g1[3] = (int)(32u << 16);        // dim1 hi=0 | tile_dim0=32
  g1[4] = 1024;                    // tile_dim1=1024, tile_dim2=0
  g1[5] = 1024;                    // tensor_dim0_stride = 1024 elems
  g1[6] = 0;
  g1[7] = 0;
  i32x4 gz = {0,0,0,0};
#if __clang_major__ >= 23
  i32x8 gz8 = {0,0,0,0,0,0,0,0};
  __builtin_amdgcn_tensor_load_to_lds(g0, g1, gz, gz, gz8, 0);
#else
  __builtin_amdgcn_tensor_load_to_lds(g0, g1, gz, gz, 0);
#endif
}
#endif

// ---------------------------------------------------------------- zero scratch
__global__ void k_zero(float* out, int* presence, int* valid){
  int i = blockIdx.x*blockDim.x + threadIdx.x;
  int stride = gridDim.x*blockDim.x;
  for (int j=i; j<NL*5; j+=stride) out[j] = 0.f;
  for (int j=i; j<SENT; j+=stride) presence[j] = 0;
  for (int j=i; j<NL;   j+=stride) valid[j] = 0;
}

// ----------------------------------------- fp32 -> bf16 weight pre-conversion
__global__ void k_cvtw(const float* __restrict__ w1, const float* __restrict__ w2,
                       __bf16* __restrict__ w1b, __bf16* __restrict__ w2b){
  int i = blockIdx.x*blockDim.x + threadIdx.x;
  int stride = gridDim.x*blockDim.x;
  for (int j=i; j<1024*1024; j+=stride){
    w1b[j] = (__bf16)w1[j];
    w2b[j] = (__bf16)w2[j];
  }
}

// ------------------------------------------------- heatmap decode + HAFM lines
__global__ void k_decode(const float* __restrict__ hm, float* __restrict__ lines,
                         float* __restrict__ jloc, float* __restrict__ joff){
  int p = blockIdx.x*blockDim.x + threadIdx.x;
  if (p >= HWD) return;
  float x0 = (float)(p % Wd), y0 = (float)(p / Wd);
  float a0   = sigmoidf(hm[0*HWD+p]);
  float a1   = sigmoidf(hm[1*HWD+p]);
  float a2   = sigmoidf(hm[2*HWD+p]);
  float dist = sigmoidf(hm[3*HWD+p]);
  float bias = sigmoidf(hm[4*HWD+p]);
  jloc[p]      = sigmoidf(hm[6*HWD+p] - hm[5*HWD+p]);   // softmax ch1 of {5,6}
  joff[p]      = sigmoidf(hm[7*HWD+p]);
  joff[HWD+p]  = sigmoidf(hm[8*HWD+p]);
  float ang0 = (a0 - 0.5f) * 2.f * PI_F;
  float c0 = cosf(ang0), s0 = sinf(ang0);
  float t1 = tanf( a1 * PI_F * 0.5f);
  float t2 = tanf(-a2 * PI_F * 0.5f);
  #pragma unroll
  for (int b=0; b<3; b++){
    float d = dist + bias*(float)(b-1);
    d = fminf(fmaxf(d, 0.f), 1.f);
    float X1 = fminf(fmaxf((c0 - s0*t1)*d*5.f + x0, 0.f), 127.f);
    float Y1 = fminf(fmaxf((s0 + c0*t1)*d*5.f + y0, 0.f), 127.f);
    float X2 = fminf(fmaxf((c0 - s0*t2)*d*5.f + x0, 0.f), 127.f);
    float Y2 = fminf(fmaxf((s0 + c0*t2)*d*5.f + y0, 0.f), 127.f);
    size_t o = (size_t)(b*HWD + p) * 4;
    lines[o+0]=X1; lines[o+1]=Y1; lines[o+2]=X2; lines[o+3]=Y2;
  }
}

// ---------------------------------------------------------------- 3x3 NMS
__global__ void k_nms(const float* __restrict__ jloc, float* __restrict__ nmsb){
  int p = blockIdx.x*blockDim.x + threadIdx.x;
  if (p >= HWD) return;
  int x = p % Wd, y = p / Wd;
  float m = -INFINITY;
  for (int dy=-1; dy<=1; dy++){
    int yy = y+dy; if (yy < 0 || yy >= Hd) continue;
    for (int dx=-1; dx<=1; dx++){
      int xx = x+dx; if (xx < 0 || xx >= Wd) continue;
      m = fmaxf(m, jloc[yy*Wd+xx]);
    }
  }
  float v = jloc[p];
  nmsb[p] = (v == m) ? v : 0.f;
}

// ------------------------------------------- iterative top-K (exact top_k sem)
__global__ void k_topk(const float* __restrict__ nmsb, const float* __restrict__ joff,
                       float* __restrict__ junc){
  __shared__ float rv[1024];
  __shared__ int   ri[1024];
  int tid = threadIdx.x;
  float lv[16];
  #pragma unroll
  for (int j=0; j<16; j++) lv[j] = nmsb[tid*16 + j];
  for (int k=0; k<KJ; k++){
    float bv = -INFINITY; int bi = tid*16;
    #pragma unroll
    for (int j=0; j<16; j++){ float v = lv[j]; if (v > bv){ bv = v; bi = tid*16 + j; } }
    rv[tid] = bv; ri[tid] = bi;
    __syncthreads();
    for (int off=512; off>0; off>>=1){
      if (tid < off){
        float ov = rv[tid+off]; int oi = ri[tid+off];
        if (ov > rv[tid] || (ov == rv[tid] && oi < ri[tid])){ rv[tid]=ov; ri[tid]=oi; }
      }
      __syncthreads();
    }
    int sel = ri[0]; float sv = rv[0];
    if (tid == 0){
      bool ok = sv > 0.008f;
      float jx = (float)(sel % Wd) + joff[sel];
      float jy = (float)(sel / Wd) + joff[HWD + sel];
      junc[2*k+0] = ok ? jx : 1e6f;
      junc[2*k+1] = ok ? jy : 1e6f;
    }
    if ((sel >> 4) == tid) lv[sel & 15] = -INFINITY;
    __syncthreads();
  }
}

// ----------------------------------- endpoint->junction argmin + id presence
__global__ void k_match(const float* __restrict__ lines, const float* __restrict__ junc,
                        int* __restrict__ presence){
  __shared__ float sj[2*KJ];
  int tid = threadIdx.x;
  for (int i=tid; i<2*KJ; i+=blockDim.x) sj[i] = junc[i];
  __syncthreads();
  int i = blockIdx.x*blockDim.x + tid;
  if (i >= NL) return;
  float X1=lines[i*4+0], Y1=lines[i*4+1], X2=lines[i*4+2], Y2=lines[i*4+3];
  float d1=INFINITY, d2=INFINITY; int j1=0, j2=0;
  for (int k=0; k<KJ; k++){
    float jx = sj[2*k], jy = sj[2*k+1];
    float ax = jx-X1, ay = jy-Y1, bx = jx-X2, by = jy-Y2;
    float da = ax*ax + ay*ay, db = bx*bx + by*by;
    if (da < d1){ d1 = da; j1 = k; }
    if (db < d2){ d2 = db; j2 = k; }
  }
  int lo = (j1 < j2) ? j1 : j2;
  int hi = (j1 < j2) ? j2 : j1;
  if (lo < hi) presence[lo*KJ + hi] = 1;
}

// ------------------------ exclusive prefix scan over id space (= sorted unique)
__global__ void k_scan(const int* __restrict__ presence, int* __restrict__ rank){
  __shared__ int sc[1024];
  __shared__ int carry;
  int tid = threadIdx.x;
  if (tid == 0) carry = 0;
  __syncthreads();
  for (int chunk=0; chunk<88; chunk++){
    int i = chunk*1024 + tid;
    int p = (i < SENT) ? presence[i] : 0;
    sc[tid] = p;
    __syncthreads();
    for (int off=1; off<1024; off<<=1){
      int add = (tid >= off) ? sc[tid-off] : 0;
      __syncthreads();
      sc[tid] += add;
      __syncthreads();
    }
    int tot = sc[1023];
    if (i < SENT) rank[i] = sc[tid] - p + carry;
    __syncthreads();
    if (tid == 0) carry += tot;
    __syncthreads();
  }
}

// -------------------------------------- scatter unique ids -> line_coords rows
__global__ void k_scatter(const int* __restrict__ presence, const int* __restrict__ rank,
                          const float* __restrict__ junc, float* __restrict__ out,
                          int* __restrict__ valid){
  int id = blockIdx.x*blockDim.x + threadIdx.x;
  if (id >= SENT || !presence[id]) return;
  int r = rank[id];
  int lo = id / KJ, hi = id % KJ;
  out[r*4+0] = junc[2*lo];
  out[r*4+1] = junc[2*lo+1];
  out[r*4+2] = junc[2*hi];
  out[r*4+3] = junc[2*hi+1];
  valid[r] = 1;
}

// --------------------------- LOI 1x1 conv: loi_t[p][o] = fc1_w @ features + b
// WMMA bf16: M=128 (out ch), N=16384 (pixels), K=256. One 16x16 tile per wave.
__global__ void k_loi(const float* __restrict__ fc1_w, const float* __restrict__ fc1_b,
                      const float* __restrict__ feat, float* __restrict__ loi_t){
  int wave = (blockIdx.x*blockDim.x + threadIdx.x) >> 5;   // 0..8191
  int lane = threadIdx.x & 31;
  int hh = lane >> 4, lm = lane & 15;
  int m0 = (wave & 7) * 16;        // out-channel tile
  int p0 = (wave >> 3) * 16;       // pixel tile
  v8f acc = {0.f,0.f,0.f,0.f,0.f,0.f,0.f,0.f};
  for (int k0=0; k0<256; k0+=32){
    v16bf a, b;
    #pragma unroll
    for (int e=0; e<16; e++){
      int Ka = e + ((e >= 8) ? 8 : 0) + 8*hh;               // A 16-bit layout
      a[e] = (__bf16)fc1_w[(m0+lm)*256 + k0 + Ka];
      int Kb = e + ((e >= 8) ? 8 : 0) + 8*hh;               // B 16-bit layout
      b[e] = (__bf16)feat[(size_t)(k0+Kb)*HWD + p0 + lm];
    }
    acc = __builtin_amdgcn_wmma_f32_16x16x32_bf16(false, a, false, b, (short)0, acc, false, false);
  }
  #pragma unroll
  for (int r=0; r<8; r++){
    int m = m0 + r + 8*hh;                                  // C/D layout
    int n = p0 + lm;
    loi_t[(size_t)n*128 + m] = acc[r] + fc1_b[m];
  }
}

// ---- fused: bilinear LOI sampling + maxpool4 + two 1024x1024 layers + head.
// One block = 16 lines, 128 threads = 4 waves. Activations stay in LDS (bf16).
// Weight K-slabs (32x1024 bf16 = 64KB) are DMA'd into LDS by the TDM
// (tensor_load_to_lds, TENSORcnt), double-buffered to overlap with WMMA.
// Each wave holds 16 accumulator tiles -> 16 back-to-back v_wmma per K-step.
__global__ void k_mlp(const float* __restrict__ loi_t,
                      const __bf16* __restrict__ W1b, const float* __restrict__ B1,
                      const __bf16* __restrict__ W2b, const float* __restrict__ B2,
                      const float* __restrict__ W3, const float* __restrict__ B3,
                      const int* __restrict__ validArr, float* __restrict__ out){
  extern __shared__ char smem[];
  __bf16* sA  = (__bf16*)(smem);                  // [16][1024] activations  32KB
  __bf16* sB  = (__bf16*)(smem + 32768);          // [16][1024] activations  32KB
  __bf16* sW0 = (__bf16*)(smem + 65536);          // weight slab buf0        64KB
  __bf16* sW1 = (__bf16*)(smem + 131072);         // weight slab buf1        64KB
  float*  lc  = (float*)(smem + 196608);          // 64 floats
  float*  red = (float*)(smem + 196608 + 256);    // 128 floats
  int*    lvl = (int*)  (smem + 196608 + 256 + 512); // 16 ints

  int tid  = threadIdx.x;
  int base = blockIdx.x * 16;
  if (tid < 64) lc[tid]  = out[base*4 + tid];
  if (tid < 16) lvl[tid] = validArr[base + tid];
  __syncthreads();

  // ---- Stage A: sample 32 steps x 128ch, maxpool k=4 -> sA[n][c*8+g]
  int c = tid;                                    // one channel per thread
  for (int n=0; n<16; n++){
    float X1=lc[n*4], Y1=lc[n*4+1], X2=lc[n*4+2], Y2=lc[n*4+3];
    #pragma unroll
    for (int g=0; g<8; g++){
      float mx = -INFINITY;
      #pragma unroll
      for (int ss=0; ss<4; ss++){
        float t  = (float)(g*4+ss) * (1.f/31.f);
        float px = fminf(fmaxf(X1*t + X2*(1.f-t), 0.f), 127.f);
        float py = fminf(fmaxf(Y1*t + Y2*(1.f-t), 0.f), 127.f);
        float fx = floorf(px), fy = floorf(py);
        int x0i = (int)fx, y0i = (int)fy;
        int x1i = (x0i+1 < 127) ? x0i+1 : 127;
        int y1i = (y0i+1 < 127) ? y0i+1 : 127;
        float wx = px - fx, wy = py - fy;
        float v00 = loi_t[((size_t)(y0i*Wd + x0i))*128 + c];
        float v01 = loi_t[((size_t)(y0i*Wd + x1i))*128 + c];
        float v10 = loi_t[((size_t)(y1i*Wd + x0i))*128 + c];
        float v11 = loi_t[((size_t)(y1i*Wd + x1i))*128 + c];
        float v = v00*(1.f-wx)*(1.f-wy) + v01*wx*(1.f-wy)
                + v10*(1.f-wx)*wy       + v11*wx*wy;
        mx = fmaxf(mx, v);
      }
      sA[n*1024 + c*8 + g] = (__bf16)mx;
    }
  }
  __syncthreads();

  int lane = tid & 31, wave = tid >> 5;
  int hh = lane >> 4, lm = lane & 15;

  // ---- Stage B/C: two 1024->1024 relu layers.
  #pragma unroll 1
  for (int layer=0; layer<2; layer++){
    const __bf16* wb = layer ? W2b : W1b;
    const float*  Bv = layer ? B2  : B1;
    __bf16* src = layer ? sB : sA;
    __bf16* dst = layer ? sA : sB;

    // prologue: slab k0=0 -> sW0
#if HAVE_TDM
    if (tid < 32){
      tdm_slab_load((unsigned)(size_t)sW0, wb);
      __builtin_amdgcn_s_wait_tensorcnt(0);
    }
#else
    for (int i = tid; i < 4096; i += 128){
      int rrow = i >> 2, q = i & 3;
      ((uint4*)sW0)[i] = ((const uint4*)(wb + (size_t)rrow*1024))[q];
    }
#endif
    __syncthreads();

    v8f zero8 = {0.f,0.f,0.f,0.f,0.f,0.f,0.f,0.f};
    v8f acc[16];
    #pragma unroll
    for (int t=0; t<16; t++) acc[t] = zero8;

    #pragma unroll 1
    for (int k0=0; k0<1024; k0+=32){
      __bf16* sWc = (k0 & 32) ? sW1 : sW0;
      __bf16* sWn = (k0 & 32) ? sW0 : sW1;
      if (k0 + 32 < 1024){
#if HAVE_TDM
        if (tid < 32) tdm_slab_load((unsigned)(size_t)sWn, wb + (size_t)(k0 + 32));
#else
        for (int i = tid; i < 4096; i += 128){
          int rrow = i >> 2, q = i & 3;
          ((uint4*)sWn)[i] = ((const uint4*)(wb + (size_t)rrow*1024 + k0 + 32))[q];
        }
#endif
      }
      // A fragment (shared by all 16 tiles of this wave): two ds_load_b128
      Frag16 af;
      af.q[0] = *(const uint4*)(src + lm*1024 + k0 + 8*hh);
      af.q[1] = *(const uint4*)(src + lm*1024 + k0 + 16 + 8*hh);
      #pragma unroll
      for (int t=0; t<16; t++){
        int n0 = wave*256 + t*16;
        Frag16 bfm;
        bfm.q[0] = *(const uint4*)(sWc + (n0+lm)*32 + 8*hh);
        bfm.q[1] = *(const uint4*)(sWc + (n0+lm)*32 + 16 + 8*hh);
        acc[t] = __builtin_amdgcn_wmma_f32_16x16x32_bf16(false, af.v, false, bfm.v,
                                                         (short)0, acc[t], false, false);
      }
#if HAVE_TDM
      if (tid < 32) __builtin_amdgcn_s_wait_tensorcnt(0);
#endif
      __syncthreads();
    }

    // epilogue: bias + relu, bf16 back to LDS
    #pragma unroll
    for (int t=0; t<16; t++){
      int n0 = wave*256 + t*16;
      #pragma unroll
      for (int r=0; r<8; r++){
        int line = r + 8*hh;                       // C/D layout
        int f = n0 + lm;
        dst[line*1024 + f] = (__bf16)fmaxf(acc[t][r] + Bv[f], 0.f);
      }
    }
    __syncthreads();
  }

  // ---- Stage D: logits = h2 . w3 + b3, sigmoid, mask by validity.
  int nline = tid >> 3, part = tid & 7;
  float s = 0.f;
  int i0 = part * 128;
  for (int i=0; i<128; i++) s += (float)sA[nline*1024 + i0 + i] * W3[i0 + i];
  red[tid] = s;
  __syncthreads();
  if (part == 0){
    float tot = 0.f;
    #pragma unroll
    for (int q=0; q<8; q++) tot += red[nline*8 + q];
    float sc = 1.f/(1.f + expf(-(tot + B3[0])));
    out[4*NL + base + nline] = lvl[nline] ? sc : 0.f;
  }
}

// ---------------------------------------------------------------------------
extern "C" void kernel_launch(void* const* d_in, const int* in_sizes, int n_in,
                              void* d_out, int out_size, void* d_ws, size_t ws_size,
                              hipStream_t stream) {
  (void)in_sizes; (void)n_in; (void)out_size; (void)ws_size;
  const float* features = (const float*)d_in[0];
  const float* heatmaps = (const float*)d_in[1];
  const float* fc1_w    = (const float*)d_in[2];
  const float* fc1_b    = (const float*)d_in[3];
  const float* w1       = (const float*)d_in[4];
  const float* b1       = (const float*)d_in[5];
  const float* w2       = (const float*)d_in[6];
  const float* b2       = (const float*)d_in[7];
  const float* w3       = (const float*)d_in[8];
  const float* b3       = (const float*)d_in[9];
  float* out = (float*)d_out;

  char* ws = (char*)d_ws;
  size_t off = 0;
  auto take = [&](size_t bytes)->char* {
    char* p = ws + off;
    off = (off + bytes + 255) & ~(size_t)255;
    return p;
  };
  float*  lines    = (float*) take((size_t)NL*4*sizeof(float));      // 786 KB
  float*  jloc     = (float*) take((size_t)HWD*sizeof(float));       // 64 KB
  float*  joff     = (float*) take((size_t)2*HWD*sizeof(float));     // 128 KB
  float*  nmsb     = (float*) take((size_t)HWD*sizeof(float));       // 64 KB
  float*  junc     = (float*) take((size_t)2*KJ*sizeof(float));      // 2.4 KB
  int*    presence = (int*)   take((size_t)SENT*sizeof(int));        // 352 KB
  int*    rank     = (int*)   take((size_t)SENT*sizeof(int));        // 352 KB
  int*    validArr = (int*)   take((size_t)NL*sizeof(int));          // 192 KB
  float*  loi_t    = (float*) take((size_t)HWD*128*sizeof(float));   // 8 MB
  __bf16* w1b      = (__bf16*)take((size_t)1024*1024*2);             // 2 MB
  __bf16* w2b      = (__bf16*)take((size_t)1024*1024*2);             // 2 MB

  k_zero   <<<512, 256, 0, stream>>>(out, presence, validArr);
  k_cvtw   <<<1024, 256, 0, stream>>>(w1, w2, w1b, w2b);
  k_decode <<<HWD/256, 256, 0, stream>>>(heatmaps, lines, jloc, joff);
  k_nms    <<<HWD/256, 256, 0, stream>>>(jloc, nmsb);
  k_topk   <<<1, 1024, 0, stream>>>(nmsb, joff, junc);
  k_match  <<<NL/256, 256, 0, stream>>>(lines, junc, presence);
  k_scan   <<<1, 1024, 0, stream>>>(presence, rank);
  k_scatter<<<(SENT+255)/256, 256, 0, stream>>>(presence, rank, junc, out, validArr);
  k_loi    <<<2048, 128, 0, stream>>>(fc1_w, fc1_b, features, loi_t);

  // sA+sB (64KB) + two weight slabs (128KB) + coords/reduction/valid
  size_t smem = 196608 + 256 + 512 + 64;
  k_mlp    <<<NL/16, 128, smem, stream>>>(loi_t, w1b, b1, w2b, b2, w3, b3, validArr, out);
}
